// MultiHeadAttention_31593779429738
// MI455X (gfx1250) — compile-verified, matrix operand order checked by reading, not run on hardware
//
#include <hip/hip_runtime.h>
#include <hip/hip_bf16.h>

// ---- types -----------------------------------------------------------------
typedef __bf16 bf16_t;
typedef __bf16 v16bf __attribute__((ext_vector_type(16)));
typedef float  v8f   __attribute__((ext_vector_type(8)));
typedef float  f32x4 __attribute__((ext_vector_type(4)));
typedef unsigned int u32x4 __attribute__((ext_vector_type(4)));
typedef int i32x4 __attribute__((ext_vector_type(4)));
typedef int i32x8 __attribute__((ext_vector_type(8)));

union Frag { v16bf v; f32x4 f[2]; };

// Problem constants
#define BB 2
#define SS 2048
#define EE 1024
#define HH 16
#define DD 64
#define MM (BB*SS)        // 4096 rows
#define HD (HH*DD)        // 1024
#define KDIM 1024         // K of both GEMMs

#if __has_builtin(__builtin_amdgcn_tensor_load_to_lds)
#define USE_TDM 1
#else
#define USE_TDM 0
#endif

__device__ __forceinline__ bf16_t f2bf(float f) {
    unsigned u = __builtin_bit_cast(unsigned, f);
    unsigned r = (u + 0x7FFFu + ((u >> 16) & 1u)) >> 16;
    unsigned short h = (unsigned short)r;
    return __builtin_bit_cast(bf16_t, h);
}

__device__ __forceinline__ v8f wmma_bf16(const Frag& a, const Frag& b, v8f c) {
    return __builtin_amdgcn_wmma_f32_16x16x32_bf16(
        false, a.v, false, b.v, (short)0, c, false, false);
}

__device__ __forceinline__ void ldfrag(Frag& fr, const bf16_t* row, int off) {
    fr.f[0] = *(const f32x4*)(row + off);
    fr.f[1] = *(const f32x4*)(row + off + 16);
}

#if USE_TDM
// TDM: load a [64 rows x 32 cols] bf16 tile from a row-major [64][2048] tensor
// into LDS at byte offset lds_addr.  D# per CDNA5 ISA ch.8 (2-D tile, 2B elems).
__device__ __forceinline__ void tdm_load_v_tile(const bf16_t* gsrc, unsigned lds_addr)
{
    unsigned long long ga = (unsigned long long)(size_t)gsrc;
    u32x4 g0;
    g0[0] = 1u;                                               // count=1, user mode
    g0[1] = lds_addr;                                         // lds_addr [63:32]
    g0[2] = (unsigned)(ga & 0xFFFFFFFFull);                   // global_addr lo
    g0[3] = (unsigned)((ga >> 32) & 0x1FFFFFFull) | 0x80000000u; // addr hi | type=2
    i32x8 g1;
    g1[0] = 0x00010000;            // workgroup_mask=0, data_size=1 (2 bytes)
    g1[1] = (int)(2048u << 16);    // tensor_dim0 = 2048 (low 16 bits at [31:16])
    g1[2] = (int)(64u   << 16);    // tensor_dim0 hi=0 | tensor_dim1 = 64
    g1[3] = (int)(32u   << 16);    // tensor_dim1 hi=0 | tile_dim0 = 32
    g1[4] = 64;                    // tile_dim1 = 64, tile_dim2 = 0 (2-D)
    g1[5] = 2048;                  // tensor_dim0_stride lo
    g1[6] = 0;                     // stride hi / dim1_stride lo
    g1[7] = 0;
    i32x4 z4 = {0, 0, 0, 0};
#if __clang_major__ >= 23
    i32x8 z8 = {0, 0, 0, 0, 0, 0, 0, 0};
    __builtin_amdgcn_tensor_load_to_lds(g0, g1, z4, z4, z8, 0);
#else
    __builtin_amdgcn_tensor_load_to_lds(g0, g1, z4, z4, 0);
#endif
}
#endif

// ---- packing kernels -------------------------------------------------------
__global__ void k_pack_x(const float* __restrict__ x, bf16_t* __restrict__ xb) {
    int i = (blockIdx.x * blockDim.x + threadIdx.x) * 4;
    xb[i + 0] = f2bf(x[i + 0]);
    xb[i + 1] = f2bf(x[i + 1]);
    xb[i + 2] = f2bf(x[i + 2]);
    xb[i + 3] = f2bf(x[i + 3]);
}

// wq (H,E,D) -> Wt[n][e] with n = h*64+d   (row-major [1024][1024])
__global__ void k_pack_w(const float* __restrict__ w, bf16_t* __restrict__ wt) {
    int i = blockIdx.x * blockDim.x + threadIdx.x;
    int n = i >> 10, e = i & 1023;
    int h = n >> 6,  d = n & 63;
    wt[i] = f2bf(w[(((size_t)h << 10) + e) * 64 + d]);
}

// wo (HD,E) -> Wot[n=e_out][k=hd]
__global__ void k_pack_wo(const float* __restrict__ wo, bf16_t* __restrict__ wot) {
    int i = blockIdx.x * blockDim.x + threadIdx.x;
    int n = i >> 10, k = i & 1023;
    wot[i] = f2bf(wo[((size_t)k << 10) + n]);
}

// ---- generic WMMA GEMM (wave tile 32x64, double-buffered K loop) -----------
// C[M][N] = A[M][K](bf16 rm) * Bt[N][K](bf16 rm)^T + bias[N]
// mode 0/1: out bf16 [b][h][s][d];  mode 2: out bf16 [b][h][d][s];  mode 3: f32
__global__ __launch_bounds__(256)
void k_gemm(const bf16_t* __restrict__ A, const bf16_t* __restrict__ Bt,
            const float* __restrict__ bias, bf16_t* __restrict__ outB,
            float* __restrict__ outF, int mode)
{
    const int lane = threadIdx.x & 31;
    const int wave = threadIdx.x >> 5;
    const int col  = lane & 15;
    const int half = lane >> 4;
    const int kb   = half * 8;
    const int mBase = blockIdx.y * 64 + (wave & 1) * 32;
    const int nBase = blockIdx.x * 256 + (wave >> 1) * 64;

    const bf16_t* arow[2];
    arow[0] = A + (size_t)(mBase + col) * KDIM;
    arow[1] = A + (size_t)(mBase + 16 + col) * KDIM;
    const bf16_t* brow[4];
#pragma unroll
    for (int nn = 0; nn < 4; ++nn)
        brow[nn] = Bt + (size_t)(nBase + nn * 16 + col) * KDIM;

    v8f acc[2][4] = {};
    Frag aP[2], bP[4], aQ[2], bQ[4];
#pragma unroll
    for (int mm = 0; mm < 2; ++mm) ldfrag(aP[mm], arow[mm], kb);
#pragma unroll
    for (int nn = 0; nn < 4; ++nn) ldfrag(bP[nn], brow[nn], kb);

    for (int k0 = 0; k0 < KDIM; k0 += 64) {
        // prefetch k0+32 while computing k0
#pragma unroll
        for (int mm = 0; mm < 2; ++mm) ldfrag(aQ[mm], arow[mm], k0 + 32 + kb);
#pragma unroll
        for (int nn = 0; nn < 4; ++nn) ldfrag(bQ[nn], brow[nn], k0 + 32 + kb);
#pragma unroll
        for (int mm = 0; mm < 2; ++mm)
#pragma unroll
            for (int nn = 0; nn < 4; ++nn)
                acc[mm][nn] = wmma_bf16(aP[mm], bP[nn], acc[mm][nn]);
        // prefetch k0+64 (over-reads 64B past K on last iter; ws-padded)
#pragma unroll
        for (int mm = 0; mm < 2; ++mm) ldfrag(aP[mm], arow[mm], k0 + 64 + kb);
#pragma unroll
        for (int nn = 0; nn < 4; ++nn) ldfrag(bP[nn], brow[nn], k0 + 64 + kb);
#pragma unroll
        for (int mm = 0; mm < 2; ++mm)
#pragma unroll
            for (int nn = 0; nn < 4; ++nn)
                acc[mm][nn] = wmma_bf16(aQ[mm], bQ[nn], acc[mm][nn]);
    }

#pragma unroll
    for (int mm = 0; mm < 2; ++mm) {
#pragma unroll
        for (int nn = 0; nn < 4; ++nn) {
            int n = nBase + nn * 16 + col;
            float bv = bias[n];
#pragma unroll
            for (int r = 0; r < 8; ++r) {
                int row = mBase + mm * 16 + half * 8 + r;
                float v = acc[mm][nn][r] + bv;
                if (mode == 3) {
                    outF[(size_t)row * EE + n] = v;
                } else {
                    int b = row >> 11, s = row & (SS - 1);
                    int h = n >> 6,  d = n & 63;
                    if (mode == 2)
                        outB[(((size_t)b * HH + h) * DD + d) * SS + s] = f2bf(v);
                    else
                        outB[(((size_t)b * HH + h) * SS + s) * DD + d] = f2bf(v);
                }
            }
        }
    }
}

// ---- flash attention -------------------------------------------------------
// One wave per 16-row Q tile of one (b,h).  D=64, S=2048.
__global__ __launch_bounds__(256)
void k_attn(const bf16_t* __restrict__ Q, const bf16_t* __restrict__ K,
            const bf16_t* __restrict__ Vt, const int* __restrict__ mask,
            bf16_t* __restrict__ Ctx)
{
    __shared__ __align__(16) bf16_t pbuf[8][16 * 32];
#if USE_TDM
    __shared__ __align__(16) bf16_t vlds[8][64 * 32];   // 32 KiB: per-wave V tile
#endif
    const int lane = threadIdx.x & 31;
    const int wave = threadIdx.x >> 5;
    const int col  = lane & 15;
    const int half = lane >> 4;
    const int kb   = half * 8;

    int gw = blockIdx.x * 8 + wave;          // 0..4095
    int bh = gw >> 7;
    int it = gw & 127;
    int b  = bh >> 4;
    int h  = bh & 15;

    const bf16_t* qptr = Q  + (((size_t)bh * SS) + it * 16) * DD;
    const bf16_t* kptr = K  + (size_t)bh * SS * DD;
    const bf16_t* vptr = Vt + (size_t)bh * DD * SS;      // [64][2048]
    const int* mrow_g  = mask + (size_t)b * SS;

    // Q A-fragments (16x64 split into two 16x32)
    Frag qf0, qf1;
    const bf16_t* qrow = qptr + col * DD;
    ldfrag(qf0, qrow, kb);
    ldfrag(qf1, qrow, 32 + kb);

    float mr[8], lr[8];
    v8f o[4] = {};
#pragma unroll
    for (int r = 0; r < 8; ++r) { mr[r] = -1e30f; lr[r] = 0.f; }

    bf16_t* pl = &pbuf[wave][0];
#if USE_TDM
    bf16_t* vl = &vlds[wave][0];
    const unsigned vl_off = (unsigned)(size_t)vl;   // LDS aperture: low 32 bits
#endif
    const float scale = 0.125f;   // 1/sqrt(64)

    for (int j = 0; j < SS; j += 32) {
#if USE_TDM
        // async DMA of the 64x32 V tile; overlaps scores + softmax
        tdm_load_v_tile(vptr + j, vl_off);
#endif
        if (j + 32 < SS)
            __builtin_prefetch(kptr + (size_t)(j + 32 + col) * DD, 0, 1);

        // batch-load all four K fragments for the two 16-col score tiles
        Frag kf[4];
        {
            const bf16_t* krow0 = kptr + (size_t)(j + col) * DD;
            const bf16_t* krow1 = kptr + (size_t)(j + 16 + col) * DD;
            ldfrag(kf[0], krow0, kb);
            ldfrag(kf[1], krow0, 32 + kb);
            ldfrag(kf[2], krow1, kb);
            ldfrag(kf[3], krow1, 32 + kb);
        }
        v8f s0 = {}, s1 = {};
        s0 = wmma_bf16(qf0, kf[0], s0);
        s0 = wmma_bf16(qf1, kf[1], s0);
        s1 = wmma_bf16(qf0, kf[2], s1);
        s1 = wmma_bf16(qf1, kf[3], s1);

        int mv0 = mrow_g[j + col];
        int mv1 = mrow_g[j + 16 + col];
        float p[2][8];
#pragma unroll
        for (int r = 0; r < 8; ++r) {
            p[0][r] = (mv0 != 0) ? s0[r] * scale : -1e30f;
            p[1][r] = (mv1 != 0) ? s1[r] * scale : -1e30f;
        }

        // online softmax: row = half*8 + r; reduce across 16 lanes in half
#pragma unroll
        for (int r = 0; r < 8; ++r) {
            float mx = fmaxf(p[0][r], p[1][r]);
            mx = fmaxf(mx, __shfl_xor(mx, 1));
            mx = fmaxf(mx, __shfl_xor(mx, 2));
            mx = fmaxf(mx, __shfl_xor(mx, 4));
            mx = fmaxf(mx, __shfl_xor(mx, 8));
            float mnew  = fmaxf(mr[r], mx);
            float alpha = __expf(mr[r] - mnew);
            p[0][r] = __expf(p[0][r] - mnew);
            p[1][r] = __expf(p[1][r] - mnew);
            float rs = p[0][r] + p[1][r];
            rs += __shfl_xor(rs, 1);
            rs += __shfl_xor(rs, 2);
            rs += __shfl_xor(rs, 4);
            rs += __shfl_xor(rs, 8);
            lr[r] = lr[r] * alpha + rs;
            mr[r] = mnew;
            o[0][r] *= alpha; o[1][r] *= alpha;
            o[2][r] *= alpha; o[3][r] *= alpha;
        }

        // C-layout -> A-layout relayout of P through LDS (bf16)
#pragma unroll
        for (int r = 0; r < 8; ++r) {
            pl[(half * 8 + r) * 32 + col]      = f2bf(p[0][r]);
            pl[(half * 8 + r) * 32 + 16 + col] = f2bf(p[1][r]);
        }
        asm volatile("s_wait_dscnt 0x0" ::: "memory");
        Frag pf;
        pf.f[0] = *(const f32x4*)(pl + col * 32 + kb);
        pf.f[1] = *(const f32x4*)(pl + col * 32 + 16 + kb);

#if USE_TDM
        __builtin_amdgcn_s_wait_tensorcnt(0);
        asm volatile("" ::: "memory");
#endif
        // o += P (16x32) * V (32x64)
#pragma unroll
        for (int nn = 0; nn < 4; ++nn) {
            Frag vf;
#if USE_TDM
            const bf16_t* vrow = vl + (size_t)(nn * 16 + col) * 32;
            vf.f[0] = *(const f32x4*)(vrow + kb);
            vf.f[1] = *(const f32x4*)(vrow + 16 + kb);
#else
            const bf16_t* vrow = vptr + (size_t)(nn * 16 + col) * SS + j;
            vf.f[0] = *(const f32x4*)(vrow + kb);
            vf.f[1] = *(const f32x4*)(vrow + 16 + kb);
#endif
            o[nn] = wmma_bf16(pf, vf, o[nn]);
        }
    }

    // normalize and write context [row][h*64+d] as bf16
    size_t rowg_base = (size_t)b * SS + it * 16;
#pragma unroll
    for (int r = 0; r < 8; ++r) {
        float inv = lr[r] > 0.f ? 1.0f / lr[r] : 0.f;
        size_t rowg = rowg_base + half * 8 + r;
#pragma unroll
        for (int nn = 0; nn < 4; ++nn) {
            int cg = h * DD + nn * 16 + col;
            Ctx[rowg * HD + cg] = f2bf(o[nn][r] * inv);
        }
    }
}

// ---- host launch -----------------------------------------------------------
extern "C" void kernel_launch(void* const* d_in, const int* in_sizes, int n_in,
                              void* d_out, int out_size, void* d_ws, size_t ws_size,
                              hipStream_t stream) {
    const float* x    = (const float*)d_in[0];
    const int*   msk  = (const int*)  d_in[1];
    const float* wq   = (const float*)d_in[2];
    const float* bq   = (const float*)d_in[3];
    const float* wk   = (const float*)d_in[4];
    const float* bk   = (const float*)d_in[5];
    const float* wv   = (const float*)d_in[6];
    const float* bv   = (const float*)d_in[7];
    const float* wo   = (const float*)d_in[8];
    const float* bo   = (const float*)d_in[9];
    float* out = (float*)d_out;

    char* ws = (char*)d_ws;
    size_t off = 0;
    auto take = [&](size_t bytes) -> char* {
        char* p = ws + off;
        off = (off + bytes + 255) & ~(size_t)255;
        return p;
    };
    // order chosen so K-loop over-reads land inside the workspace
    bf16_t* Xb  = (bf16_t*)take((size_t)MM * EE * 2);   // 8 MiB
    bf16_t* Wqt = (bf16_t*)take((size_t)HD * EE * 2);   // 2 MiB
    bf16_t* Wkt = (bf16_t*)take((size_t)HD * EE * 2);
    bf16_t* Wvt = (bf16_t*)take((size_t)HD * EE * 2);
    bf16_t* Wot = (bf16_t*)take((size_t)EE * HD * 2);
    bf16_t* Ctx = (bf16_t*)take((size_t)MM * HD * 2);   // 8 MiB
    bf16_t* Qb  = (bf16_t*)take((size_t)MM * HD * 2);
    bf16_t* Kb  = (bf16_t*)take((size_t)MM * HD * 2);
    bf16_t* Vtb = (bf16_t*)take((size_t)MM * HD * 2);

    // pack
    k_pack_x <<<(MM * EE) / (256 * 4), 256, 0, stream>>>(x, Xb);
    k_pack_w <<<(HD * EE) / 256, 256, 0, stream>>>(wq, Wqt);
    k_pack_w <<<(HD * EE) / 256, 256, 0, stream>>>(wk, Wkt);
    k_pack_w <<<(HD * EE) / 256, 256, 0, stream>>>(wv, Wvt);
    k_pack_wo<<<(EE * HD) / 256, 256, 0, stream>>>(wo, Wot);

    // projections: WG tile 64x256, wave tile 32x64
    dim3 gg(HD / 256, MM / 64);
    k_gemm<<<gg, 256, 0, stream>>>(Xb, Wqt, bq, Qb,  nullptr, 0);
    k_gemm<<<gg, 256, 0, stream>>>(Xb, Wkt, bk, Kb,  nullptr, 1);
    k_gemm<<<gg, 256, 0, stream>>>(Xb, Wvt, bv, Vtb, nullptr, 2);

    // attention: B*H*(S/16) = 4096 waves, 8 waves / block
    k_attn<<<(BB * HH * (SS / 16)) / 8, 256, 0, stream>>>(Qb, Kb, Vtb, msk, Ctx);

    // output projection (f32 out + bias)
    k_gemm<<<gg, 256, 0, stream>>>(Ctx, Wot, bo, nullptr, out, 3);
}
